// UrbanRidehailKG_68143951118751
// MI455X (gfx1250) — matrix operand order repaired and checked by reading, not compiled, not run on hardware
//
#include <hip/hip_runtime.h>
#include <hip/hip_bf16.h>
#include <math.h>

typedef _Float16 v16h __attribute__((ext_vector_type(16)));
typedef _Float16 h8   __attribute__((ext_vector_type(8)));
typedef _Float16 h4   __attribute__((ext_vector_type(4)));
typedef float    v8f  __attribute__((ext_vector_type(8)));

#define NB 65536      // batch
#define EE 512        // embedding dim
#define HH 8          // heads
#define HD 64         // head dim
#define MM 8          // motifs

__device__ __forceinline__ float wave_sum32(float v) {
#pragma unroll
  for (int s = 1; s < 32; s <<= 1) v += __shfl_xor(v, s, 32);
  return v;
}

#define WMMA_F16(A, B, C) \
  __builtin_amdgcn_wmma_f32_16x16x32_f16(false, (A), false, (B), (short)0, (C), false, false)

// ---------------------------------------------------------------------------
// Repack a K x N fp32 row-major weight matrix into wave32 WMMA B-fragments
// (f16).  Fragment (nt, kt), lane l holds column n = nt*16 + (l&15),
// K kb..kb+7 then kb+16..kb+23 where kb = kt*32 + (l<16 ? 0 : 8).  16
// halves per lane -> GEMM reads one 32B v16h per fragment.
// ---------------------------------------------------------------------------
__global__ __launch_bounds__(256) void repack_w_kernel(
    const float* __restrict__ W, _Float16* __restrict__ P, int K, int N) {
  const int KT = K / 32;
  const int total = (N / 16) * KT * 32;
  int t = blockIdx.x * 256 + threadIdx.x;
  if (t >= total) return;
  const int lane = t & 31;
  const int tile = t >> 5;              // tile = nt*KT + kt
  const int kt = tile % KT;
  const int nt = tile / KT;
  const int n  = nt * 16 + (lane & 15);
  const int kb = kt * 32 + ((lane < 16) ? 0 : 8);
  _Float16* dst = P + (size_t)t * 16;
#pragma unroll
  for (int j = 0; j < 16; ++j)
    dst[j] = (_Float16)W[(size_t)(kb + j + ((j >= 8) ? 8 : 0)) * N + n];
}

// ---------------------------------------------------------------------------
// Tiny precompute: lk = local @ Wkl + bkl, gk = global @ Wkg + bkg,
// motifs = local + global.  512 outputs each, one block.
// ---------------------------------------------------------------------------
__global__ __launch_bounds__(512) void motif_prep_kernel(
    const float* __restrict__ local_m, const float* __restrict__ global_m,
    const float* __restrict__ Wkl, const float* __restrict__ bkl,
    const float* __restrict__ Wkg, const float* __restrict__ bkg,
    float* __restrict__ lk, float* __restrict__ gk, float* __restrict__ motifs) {
  const int t = threadIdx.x;            // 0..511
  const int m = t >> 6, d = t & 63;
  float a = bkl[d], g = bkg[d];
#pragma unroll 8
  for (int k = 0; k < HD; ++k) {
    a += local_m[m * HD + k]  * Wkl[k * HD + d];
    g += global_m[m * HD + k] * Wkg[k * HD + d];
  }
  lk[t] = a; gk[t] = g;
  motifs[t] = local_m[t] + global_m[t];
}

// ---------------------------------------------------------------------------
// Gather + elementwise stage: one wave per batch element.
// Produces ctx (f16), diff (f32 + f16 into combined[:,0:512]), density.
// ---------------------------------------------------------------------------
__global__ __launch_bounds__(256) void gather_prep_kernel(
    const float* __restrict__ entity_emb, const float* __restrict__ relation_emb,
    const float* __restrict__ dens_ent_emb, const float* __restrict__ dens_ent_w,
    const float* __restrict__ dens_ent_b,  const float* __restrict__ dens_rel_emb,
    const float* __restrict__ dens_rel_w,  const float* __restrict__ dens_rel_b,
    const int* __restrict__ h_idx, const int* __restrict__ r_idx,
    const int* __restrict__ t_idx,
    _Float16* __restrict__ ctx16, _Float16* __restrict__ combined,
    float* __restrict__ diff32, float* __restrict__ density) {
  const int lane = threadIdx.x & 31;
  const int wave = threadIdx.x >> 5;
  const size_t b = (size_t)blockIdx.x * 8 + wave;
  const int hi = h_idx[b], ri = r_idx[b], ti = t_idx[b];

  const float4* H  = (const float4*)(entity_emb   + (size_t)hi * EE);
  const float4* R  = (const float4*)(relation_emb + (size_t)ri * EE);
  const float4* T  = (const float4*)(entity_emb   + (size_t)ti * EE);
  const float4* DH = (const float4*)(dens_ent_emb + (size_t)hi * EE);
  const float4* DT = (const float4*)(dens_ent_emb + (size_t)ti * EE);
  const float4* DR = (const float4*)(dens_rel_emb + (size_t)ri * EE);
  const float4* WE = (const float4*)dens_ent_w;
  const float4* WR = (const float4*)dens_rel_w;

  float4* Dout = (float4*)(diff32 + b * EE);
  _Float16* Cx = ctx16   + b * EE;
  _Float16* Cm = combined + b * (2 * EE);

  float ah = 0.f, at = 0.f, ar = 0.f;
#pragma unroll
  for (int c = 0; c < 4; ++c) {
    const int j = lane + 32 * c;        // float4 index 0..127
    const float4 hh = H[j], rr = R[j], tt = T[j];
    float4 df, cx;
    df.x = hh.x + rr.x - tt.x; df.y = hh.y + rr.y - tt.y;
    df.z = hh.z + rr.z - tt.z; df.w = hh.w + rr.w - tt.w;
    cx.x = hh.x + rr.x + tt.x; cx.y = hh.y + rr.y + tt.y;
    cx.z = hh.z + rr.z + tt.z; cx.w = hh.w + rr.w + tt.w;
    Dout[j] = df;
    h4 cf, dh16;
    cf[0] = (_Float16)cx.x; cf[1] = (_Float16)cx.y;
    cf[2] = (_Float16)cx.z; cf[3] = (_Float16)cx.w;
    dh16[0] = (_Float16)df.x; dh16[1] = (_Float16)df.y;
    dh16[2] = (_Float16)df.z; dh16[3] = (_Float16)df.w;
    *(h4*)(Cx + j * 4) = cf;
    *(h4*)(Cm + j * 4) = dh16;          // combined[:, 0:512] = diff

    const float4 we = WE[j], wr = WR[j];
    const float4 dh = DH[j], dt = DT[j], dr = DR[j];
    ah += dh.x * we.x + dh.y * we.y + dh.z * we.z + dh.w * we.w;
    at += dt.x * we.x + dt.y * we.y + dt.z * we.z + dt.w * we.w;
    ar += dr.x * wr.x + dr.y * wr.y + dr.z * wr.z + dr.w * wr.w;
  }
  ah = wave_sum32(ah); at = wave_sum32(at); ar = wave_sum32(ar);
  if (lane == 0) {
    const float be = dens_ent_b[0], br = dens_rel_b[0];
    const float sh = 1.f / (1.f + __expf(-(ah + be)));
    const float st = 1.f / (1.f + __expf(-(at + be)));
    const float sr = 1.f / (1.f + __expf(-(ar + br)));
    density[b] = (sh + st + sr) * (1.f / 3.f);
  }
}

// ---------------------------------------------------------------------------
// A-fragment loader: 16-bit A 16x32 layout (lane m = l&15; kbase 0 / 8;
// VGPR0..3 = K kbase..kbase+7, VGPR4..7 = K kbase+16..kbase+23).
// ---------------------------------------------------------------------------
__device__ __forceinline__ v16h load_afrag(const _Float16* __restrict__ Arow,
                                           int koff) {
  const h8 lo = *(const h8*)(Arow + koff);
  const h8 hi = *(const h8*)(Arow + koff + 16);
  v16h a;
#pragma unroll
  for (int i = 0; i < 8; ++i) { a[i] = lo[i]; a[i + 8] = hi[i]; }
  return a;
}

// ---------------------------------------------------------------------------
// Software-pipelined WMMA mainloop: each wave computes a 32x64 output tile
// (2 M-tiles x 4 N-tiles).  Next iteration's A/B fragments are issued before
// the current iteration's 8 WMMAs so loads overlap the matrix pipe.
// ---------------------------------------------------------------------------
__device__ __forceinline__ void gemm_mainloop_32x64(
    const _Float16* __restrict__ Arow0, const _Float16* __restrict__ Arow1,
    const _Float16* __restrict__ BpW, size_t fragStride, int KT, int kbase,
    v8f acc[2][4]) {
  v16h a0 = load_afrag(Arow0, kbase);
  v16h a1 = load_afrag(Arow1, kbase);
  v16h b0 = *(const v16h*)(BpW);
  v16h b1 = *(const v16h*)(BpW + fragStride);
  v16h b2 = *(const v16h*)(BpW + 2 * fragStride);
  v16h b3 = *(const v16h*)(BpW + 3 * fragStride);
  for (int kt = 0; kt < KT; ++kt) {
    const int ktn = (kt + 1 < KT) ? (kt + 1) : kt;   // clamp: no OOB prefetch
    // issue next-iteration loads first (stay in flight across the WMMAs)
    const v16h a0n = load_afrag(Arow0, ktn * 32 + kbase);
    const v16h a1n = load_afrag(Arow1, ktn * 32 + kbase);
    const _Float16* bk = BpW + (size_t)ktn * 512;    // 32 lanes * 16 halves
    const v16h b0n = *(const v16h*)(bk);
    const v16h b1n = *(const v16h*)(bk + fragStride);
    const v16h b2n = *(const v16h*)(bk + 2 * fragStride);
    const v16h b3n = *(const v16h*)(bk + 3 * fragStride);
    acc[0][0] = WMMA_F16(a0, b0, acc[0][0]);
    acc[0][1] = WMMA_F16(a0, b1, acc[0][1]);
    acc[0][2] = WMMA_F16(a0, b2, acc[0][2]);
    acc[0][3] = WMMA_F16(a0, b3, acc[0][3]);
    acc[1][0] = WMMA_F16(a1, b0, acc[1][0]);
    acc[1][1] = WMMA_F16(a1, b1, acc[1][1]);
    acc[1][2] = WMMA_F16(a1, b2, acc[1][2]);
    acc[1][3] = WMMA_F16(a1, b3, acc[1][3]);
    a0 = a0n; a1 = a1n; b0 = b0n; b1 = b1n; b2 = b2n; b3 = b3n;
  }
}

// ---------------------------------------------------------------------------
// q = ctx @ Wq + bq   (M=65536, N=512, K=512), f16 in / f32 acc / f16 out.
// Block = 32 rows x 512 cols (8 waves, each 32x64).
// ---------------------------------------------------------------------------
__global__ __launch_bounds__(256) void gemm_q_kernel(
    const _Float16* __restrict__ A, const _Float16* __restrict__ Bp,
    const float* __restrict__ bias, _Float16* __restrict__ Out) {
  const int K = EE, KT = K / 32, N = EE;
  const int lane = threadIdx.x & 31;
  const int wave = threadIdx.x >> 5;
  const int mblk = blockIdx.x;
  const int kbase = (lane < 16) ? 0 : 8;
  const _Float16* Arow0 = A + (size_t)(mblk * 32 + (lane & 15)) * K;
  const _Float16* Arow1 = Arow0 + (size_t)16 * K;
  const size_t fragStride = (size_t)KT * 32 * 16;       // halves per n-tile
  const _Float16* BpW = Bp + (size_t)(wave * 4) * fragStride + (size_t)lane * 16;

  v8f acc[2][4] = {};
  gemm_mainloop_32x64(Arow0, Arow1, BpW, fragStride, KT, kbase, acc);

#pragma unroll
  for (int t = 0; t < 2; ++t) {
    const int rbase = mblk * 32 + t * 16 + ((lane >= 16) ? 8 : 0);
#pragma unroll
    for (int j = 0; j < 4; ++j) {
      const int n = (wave * 4 + j) * 16 + (lane & 15);
      const float bn = bias[n];
#pragma unroll
      for (int i = 0; i < 8; ++i)
        Out[(size_t)(rbase + i) * N + n] = (_Float16)(acc[t][j][i] + bn);
    }
  }
}

// ---------------------------------------------------------------------------
// Attention over motifs, motif_repr -> combined[:,512:1024], motif_loss.
// One wave per batch element; lane owns 2 dims of HD=64.
// ---------------------------------------------------------------------------
__global__ __launch_bounds__(256) void attn_kernel(
    const _Float16* __restrict__ q, const float* __restrict__ diff,
    const float* __restrict__ density, const float* __restrict__ lk,
    const float* __restrict__ gk, const float* __restrict__ motifs,
    _Float16* __restrict__ combined, float* __restrict__ loss) {
  const int lane = threadIdx.x & 31;
  const int wave = threadIdx.x >> 5;
  const size_t b = (size_t)blockIdx.x * 8 + wave;
  const float scale = 0.125f;           // 1/sqrt(64)
  const int d0 = lane * 2;

  float lkx[MM], lky[MM], gkx[MM], gky[MM], mx[MM], my[MM];
#pragma unroll
  for (int m = 0; m < MM; ++m) {
    lkx[m] = lk[m * HD + d0];     lky[m] = lk[m * HD + d0 + 1];
    gkx[m] = gk[m * HD + d0];     gky[m] = gk[m * HD + d0 + 1];
    mx[m]  = motifs[m * HD + d0]; my[m]  = motifs[m * HD + d0 + 1];
  }
  const float dens = density[b];
  const _Float16* qb = q + b * EE;
  const float* db = diff + b * EE;
  _Float16* cb = combined + b * (2 * EE) + EE;

  float lossacc = 0.f;
#pragma unroll
  for (int h = 0; h < HH; ++h) {
    const float qx = (float)qb[h * HD + d0];
    const float qy = (float)qb[h * HD + d0 + 1];
    float al[MM], ag[MM];
#pragma unroll
    for (int m = 0; m < MM; ++m) {
      al[m] = qx * lkx[m] + qy * lky[m];
      ag[m] = qx * gkx[m] + qy * gky[m];
    }
#pragma unroll
    for (int s = 1; s < 32; s <<= 1) {
#pragma unroll
      for (int m = 0; m < MM; ++m) {
        al[m] += __shfl_xor(al[m], s, 32);
        ag[m] += __shfl_xor(ag[m], s, 32);
      }
    }
    float ml = -1e30f, mg = -1e30f;
#pragma unroll
    for (int m = 0; m < MM; ++m) {
      al[m] = al[m] * scale + dens;  ml = fmaxf(ml, al[m]);
      ag[m] = ag[m] * scale + dens;  mg = fmaxf(mg, ag[m]);
    }
    float sl = 0.f, sg = 0.f;
#pragma unroll
    for (int m = 0; m < MM; ++m) {
      al[m] = __expf(al[m] - ml); sl += al[m];
      ag[m] = __expf(ag[m] - mg); sg += ag[m];
    }
    const float il = 1.f / sl, ig = 1.f / sg;
    float repx = 0.f, repy = 0.f;
#pragma unroll
    for (int m = 0; m < MM; ++m) {
      const float attn = 0.5f * (al[m] * il + ag[m] * ig);
      repx += attn * mx[m];
      repy += attn * my[m];
    }
    const float dx = db[h * HD + d0], dy = db[h * HD + d0 + 1];
    lossacc += (dx - repx) * (dx - repx) + (dy - repy) * (dy - repy);
    cb[h * HD + d0]     = (_Float16)repx;
    cb[h * HD + d0 + 1] = (_Float16)repy;
  }
  lossacc = wave_sum32(lossacc);
  if (lane == 0) loss[b] = lossacc;
}

// ---------------------------------------------------------------------------
// transformed = tanh(combined @ Wout + bout); score = adv_w*||.||_2 + adv_b.
// Same pipelined WMMA mainloop (K=1024); row sum-of-squares fused via lane
// shuffles + LDS atomics so (B,512) transformed never touches memory.
// ---------------------------------------------------------------------------
__global__ __launch_bounds__(256) void gemm_out_score_kernel(
    const _Float16* __restrict__ A, const _Float16* __restrict__ Bp,
    const float* __restrict__ bias, const float* __restrict__ adv_w,
    const float* __restrict__ adv_b, float* __restrict__ score) {
  const int K = 2 * EE, KT = K / 32;
  const int lane = threadIdx.x & 31;
  const int wave = threadIdx.x >> 5;
  const int mblk = blockIdx.x;
  const int kbase = (lane < 16) ? 0 : 8;
  const _Float16* Arow0 = A + (size_t)(mblk * 32 + (lane & 15)) * K;
  const _Float16* Arow1 = Arow0 + (size_t)16 * K;
  const size_t fragStride = (size_t)KT * 32 * 16;
  const _Float16* BpW = Bp + (size_t)(wave * 4) * fragStride + (size_t)lane * 16;

  __shared__ float ssum[32];
  if (threadIdx.x < 32) ssum[threadIdx.x] = 0.f;
  __syncthreads();

  v8f acc[2][4] = {};
  gemm_mainloop_32x64(Arow0, Arow1, BpW, fragStride, KT, kbase, acc);

#pragma unroll
  for (int t = 0; t < 2; ++t) {
    float ss[8];
#pragma unroll
    for (int i = 0; i < 8; ++i) ss[i] = 0.f;
#pragma unroll
    for (int j = 0; j < 4; ++j) {
      const int n = (wave * 4 + j) * 16 + (lane & 15);
      const float bn = bias[n];
#pragma unroll
      for (int i = 0; i < 8; ++i) {
        const float v = tanhf(acc[t][j][i] + bn);
        ss[i] += v * v;
      }
    }
    // reduce across the 16 lanes of each half (columns of this wave's tiles)
#pragma unroll
    for (int m = 1; m < 16; m <<= 1) {
#pragma unroll
      for (int i = 0; i < 8; ++i) ss[i] += __shfl_xor(ss[i], m, 32);
    }
    if ((lane & 15) == 0) {
      const int rb = t * 16 + ((lane >= 16) ? 8 : 0);
#pragma unroll
      for (int i = 0; i < 8; ++i) atomicAdd(&ssum[rb + i], ss[i]);
    }
  }
  __syncthreads();
  if (threadIdx.x < 32) {
    score[(size_t)mblk * 32 + threadIdx.x] =
        adv_w[0] * sqrtf(ssum[threadIdx.x]) + adv_b[0];
  }
}

// ---------------------------------------------------------------------------
extern "C" void kernel_launch(void* const* d_in, const int* in_sizes, int n_in,
                              void* d_out, int out_size, void* d_ws,
                              size_t ws_size, hipStream_t stream) {
  const float* entity_emb   = (const float*)d_in[0];
  const float* relation_emb = (const float*)d_in[1];
  const float* dens_ent_emb = (const float*)d_in[2];
  const float* dens_ent_w   = (const float*)d_in[3];
  const float* dens_ent_b   = (const float*)d_in[4];
  const float* dens_rel_emb = (const float*)d_in[5];
  const float* dens_rel_w   = (const float*)d_in[6];
  const float* dens_rel_b   = (const float*)d_in[7];
  const float* local_motifs = (const float*)d_in[8];
  const float* global_motifs= (const float*)d_in[9];
  const float* Wq           = (const float*)d_in[10];
  const float* bq           = (const float*)d_in[11];
  const float* Wkl          = (const float*)d_in[12];
  const float* bkl          = (const float*)d_in[13];
  const float* Wkg          = (const float*)d_in[14];
  const float* bkg          = (const float*)d_in[15];
  const float* Wout         = (const float*)d_in[16];
  const float* bout         = (const float*)d_in[17];
  const float* adv_w        = (const float*)d_in[18];
  const float* adv_b        = (const float*)d_in[19];
  const int*   h_idx        = (const int*)d_in[20];
  const int*   r_idx        = (const int*)d_in[21];
  const int*   t_idx        = (const int*)d_in[22];

  char* ws = (char*)d_ws;
  const size_t szB = (size_t)NB;
  size_t off = 0;
  _Float16* ctx16   = (_Float16*)(ws + off); off += szB * EE * 2;        // 67 MB
  _Float16* comb16  = (_Float16*)(ws + off); off += szB * 2 * EE * 2;    // 134 MB
  _Float16* q16     = (_Float16*)(ws + off); off += szB * EE * 2;        // 67 MB
  float*    diff32  = (float*)(ws + off);    off += szB * EE * 4;        // 134 MB
  float*    density = (float*)(ws + off);    off += szB * 4;
  _Float16* wqpack  = (_Float16*)(ws + off); off += (size_t)EE * EE * 2;
  _Float16* wopack  = (_Float16*)(ws + off); off += (size_t)2 * EE * EE * 2;
  float*    lk      = (float*)(ws + off);    off += MM * HD * 4;
  float*    gk      = (float*)(ws + off);    off += MM * HD * 4;
  float*    motifs  = (float*)(ws + off);    off += MM * HD * 4;

  float* out_score = (float*)d_out;
  float* out_loss  = (float*)d_out + NB;

  // weight repack: Wq (512x512), Wout (1024x512)
  {
    const int tq = (EE / 16) * (EE / 32) * 32;       // 16384 threads
    repack_w_kernel<<<(tq + 255) / 256, 256, 0, stream>>>(Wq, wqpack, EE, EE);
    const int to = (EE / 16) * (2 * EE / 32) * 32;   // 32768 threads
    repack_w_kernel<<<(to + 255) / 256, 256, 0, stream>>>(Wout, wopack, 2 * EE, EE);
  }
  motif_prep_kernel<<<1, 512, 0, stream>>>(local_motifs, global_motifs,
                                           Wkl, bkl, Wkg, bkg, lk, gk, motifs);
  gather_prep_kernel<<<NB / 8, 256, 0, stream>>>(
      entity_emb, relation_emb, dens_ent_emb, dens_ent_w, dens_ent_b,
      dens_rel_emb, dens_rel_w, dens_rel_b, h_idx, r_idx, t_idx,
      ctx16, comb16, diff32, density);
  gemm_q_kernel<<<NB / 32, 256, 0, stream>>>(ctx16, wqpack, bq, q16);
  attn_kernel<<<NB / 8, 256, 0, stream>>>(q16, diff32, density, lk, gk, motifs,
                                          comb16, out_loss);
  gemm_out_score_kernel<<<NB / 32, 256, 0, stream>>>(comb16, wopack, bout,
                                                     adv_w, adv_b, out_score);
}
